// SwinTransformerV2StageUp_10728828305435
// MI455X (gfx1250) — matrix operand order repaired and checked by reading, not compiled, not run on hardware
//
#include <hip/hip_runtime.h>
#include <hip/hip_fp16.h>
#include <math.h>

typedef __attribute__((ext_vector_type(16))) _Float16 v16h;
typedef __attribute__((ext_vector_type(8)))  float    v8f;
typedef __attribute__((ext_vector_type(4)))  int      v4i;

#define NTOK 65536      // B*H*W tokens
#define DIM  192
#define HEADS 6
#define NWIN 1024       // B * 8 * 8 windows
#define LOG100 4.6051701859880914f

// D = A*B + C, f16 inputs, f32 accum, 16x16x32
__device__ __forceinline__ v8f wmma16(v16h a, v16h b, v8f c) {
  return __builtin_amdgcn_wmma_f32_16x16x32_f16(false, a, false, b, (short)0, c, false, false);
}

// ---- async 16B global->LDS copy (ASYNCcnt path), with safe fallback ----
__device__ __forceinline__ void async_copy16(void* lds, const void* g) {
#if __has_builtin(__builtin_amdgcn_global_load_async_to_lds_b128)
  __builtin_amdgcn_global_load_async_to_lds_b128(
      (__attribute__((address_space(1))) v4i*)g,
      (__attribute__((address_space(3))) v4i*)lds, 0, 0);
#else
  *(float4*)lds = *(const float4*)g;
#endif
}
__device__ __forceinline__ void async_wait0() {
#if __has_builtin(__builtin_amdgcn_s_wait_asynccnt)
  __builtin_amdgcn_s_wait_asynccnt(0);
#endif
}

// A fragment (16x32 at row0,k0) from row-major f16 LDS, leading dim ld.
// lane m=lane%16, half=lane/16; element j -> k=16*half+j
__device__ __forceinline__ v16h load_a_h(const _Float16* S, int ld, int row0, int k0) {
  int lane = threadIdx.x & 31;
  int m = lane & 15, half = lane >> 4;
  const _Float16* r = S + (row0 + m) * ld + k0 + 16 * half;
  v16h a;
#pragma unroll
  for (int j = 0; j < 8; ++j) a[j] = r[j];
#pragma unroll
  for (int j = 0; j < 8; ++j) a[8 + j] = r[8 + j];
  return a;
}

// A fragment from row-major f32 storage (attention S/P, q) with conversion
__device__ __forceinline__ v16h load_a_f(const float* S, int ld, int row0, int k0) {
  int lane = threadIdx.x & 31;
  int m = lane & 15, half = lane >> 4;
  const float* r = S + (row0 + m) * ld + k0 + 16 * half;
  v16h a;
#pragma unroll
  for (int j = 0; j < 8; ++j) a[j] = (_Float16)r[j];
#pragma unroll
  for (int j = 0; j < 8; ++j) a[8 + j] = (_Float16)r[8 + j];
  return a;
}

// B fragment (32x16 at k0,n0), B[k][n] = W[n][k], W row-major NxK in f16 (pre-converted weights)
// lane n=lane%16, half=lane/16; element j -> k = k0 + 16*half + j  (32 contiguous bytes)
__device__ __forceinline__ v16h load_bT_h(const _Float16* W, int ld, int k0, int n0) {
  int lane = threadIdx.x & 31;
  int n = lane & 15, half = lane >> 4;
  const _Float16* r = W + (size_t)(n0 + n) * ld + k0 + 16 * half;
  v16h b;
#pragma unroll
  for (int j = 0; j < 16; ++j) b[j] = r[j];
  return b;
}

// same but from f32 storage (k-normalized keys in LDS)
__device__ __forceinline__ v16h load_bT_f(const float* W, int ld, int k0, int n0) {
  int lane = threadIdx.x & 31;
  int n = lane & 15, half = lane >> 4;
  const float* r = W + (size_t)(n0 + n) * ld + k0 + 16 * half;
  v16h b;
#pragma unroll
  for (int j = 0; j < 16; ++j) b[j] = (_Float16)r[j];
  return b;
}

// B fragment, K-major storage: B[k][n] = S[k*ld + n]  (f32 V in LDS)
__device__ __forceinline__ v16h load_b_kmaj(const float* S, int ld, int k0, int n0) {
  int lane = threadIdx.x & 31;
  int n = lane & 15, half = lane >> 4;
  const float* p = S + (size_t)(k0 + 16 * half) * ld + n0 + n;
  v16h b;
#pragma unroll
  for (int j = 0; j < 16; ++j) b[j] = (_Float16)p[(size_t)j * ld];
  return b;
}

// ---------------- weight conversion f32 -> f16 ----------------
__global__ void k_cvt(const float* __restrict__ s, _Float16* __restrict__ d, int n) {
  int i = blockIdx.x * blockDim.x + threadIdx.x;
  if (i < n) d[i] = (_Float16)s[i];
}

// ---------------- CPB (continuous relative position bias) ----------------
__device__ __forceinline__ float relcoord(int a) {
  float v = (float)(a - 7) * (8.0f / 7.0f);
  float s = (v > 0.f) ? 1.f : ((v < 0.f) ? -1.f : 0.f);
  return s * log2f(fabsf(v) + 1.f) * (1.0f / 3.0f); // log2(8)=3
}

__global__ void k_cpb1(const float* __restrict__ w1, const float* __restrict__ b1,
                       const float* __restrict__ w2, float* __restrict__ vals) {
  int idx = blockIdx.x * blockDim.x + threadIdx.x;   // d*225*6 + t*6 + head
  if (idx >= 2 * 225 * 6) return;
  int d = idx / (225 * 6);
  int t = (idx / 6) % 225;
  int head = idx % 6;
  float t0 = relcoord(t / 15), t1 = relcoord(t % 15);
  const float* W1 = w1 + d * 512 * 2;
  const float* B1 = b1 + d * 512;
  const float* W2 = w2 + (d * 6 + head) * 512;
  float acc = 0.f;
  for (int k = 0; k < 512; ++k) {
    float hv = fmaxf(t0 * W1[2*k] + t1 * W1[2*k+1] + B1[k], 0.f);
    acc += hv * W2[k];
  }
  vals[idx] = acc;
}

__global__ void k_cpb2(const float* __restrict__ vals, float* __restrict__ biasTab) {
  int idx = blockIdx.x * blockDim.x + threadIdx.x;   // d*6*4096 + head*4096 + i*64 + j
  if (idx >= 2 * 6 * 4096) return;
  int d = idx / (6 * 4096);
  int head = (idx / 4096) % 6;
  int ij = idx & 4095;
  int i = ij >> 6, j = ij & 63;
  int dy = (i >> 3) - (j >> 3) + 7;
  int dx = (i & 7) - (j & 7) + 7;
  float v = vals[(d * 225 + dy * 15 + dx) * 6 + head];
  biasTab[idx] = 16.f / (1.f + __expf(-v));
}

// ---------------- skip fusion: concat(x,skip) @ cat_w.T + cat_b ----------------
__global__ void __launch_bounds__(256) k_fuse(const float* __restrict__ x, const float* __restrict__ skip,
                                              const _Float16* __restrict__ cat_wh, const float* __restrict__ cat_b,
                                              float* __restrict__ xA, _Float16* __restrict__ xAh) {
  __shared__ _Float16 aA[64 * 384];   // 48 KB
  int tok0 = blockIdx.x * 64;
  for (int idx = threadIdx.x; idx < 64 * 96; idx += 256) {
    int r = idx / 96, c4 = idx % 96;
    float4 v = (c4 < 48) ? ((const float4*)(x + (size_t)(tok0 + r) * 192))[c4]
                         : ((const float4*)(skip + (size_t)(tok0 + r) * 192))[c4 - 48];
    _Float16* dst = aA + r * 384 + c4 * 4;
    dst[0] = (_Float16)v.x; dst[1] = (_Float16)v.y;
    dst[2] = (_Float16)v.z; dst[3] = (_Float16)v.w;
  }
  __syncthreads();
  int wv = threadIdx.x >> 5, lane = threadIdx.x & 31, half = lane >> 4;
  for (int t = wv; t < 48; t += 8) {
    int mt = t / 12, nt = t % 12;
    v8f acc = {0.f,0.f,0.f,0.f,0.f,0.f,0.f,0.f};
#pragma unroll
    for (int k0 = 0; k0 < 384; k0 += 32)
      acc = wmma16(load_a_h(aA, 384, mt * 16, k0), load_bT_h(cat_wh, 384, k0, nt * 16), acc);
    int n = nt * 16 + (lane & 15);
    float bias = cat_b[n];
#pragma unroll
    for (int r = 0; r < 8; ++r) {
      size_t o = (size_t)(tok0 + mt * 16 + r + 8 * half) * 192 + n;
      float v = acc[r] + bias;
      xA[o] = v;
      xAh[o] = (_Float16)v;
    }
  }
}

// ---------------- QKV GEMM per window (cyclic-shift gather, async f16 staging) ----------------
__global__ void __launch_bounds__(256) k_qkv(const _Float16* __restrict__ xh, const _Float16* __restrict__ qkv_wh,
                                             const float* __restrict__ q_bias, const float* __restrict__ v_bias,
                                             _Float16* __restrict__ qkv, int shift) {
  __shared__ _Float16 aA[64 * 192];   // 24 KB
  int win = blockIdx.x;
  int b = win >> 6, hy = (win >> 3) & 7, wx = win & 7;
  for (int idx = threadIdx.x; idx < 64 * 24; idx += 256) {   // 24 x (8 f16) chunks per row
    int r = idx / 24, c8 = idx % 24;
    int h = (hy * 8 + (r >> 3) + shift) & 63;
    int w = (wx * 8 + (r & 7) + shift) & 63;
    size_t tok = (size_t)(b * 64 + h) * 64 + w;
    async_copy16(aA + r * 192 + c8 * 8, xh + tok * 192 + c8 * 8);
  }
  async_wait0();
  __syncthreads();
  int wv = threadIdx.x >> 5, lane = threadIdx.x & 31, half = lane >> 4;
  for (int t = wv; t < 144; t += 8) {
    int mt = t / 36, nt = t % 36;
    v8f acc = {0.f,0.f,0.f,0.f,0.f,0.f,0.f,0.f};
#pragma unroll
    for (int k0 = 0; k0 < 192; k0 += 32)
      acc = wmma16(load_a_h(aA, 192, mt * 16, k0), load_bT_h(qkv_wh, 192, k0, nt * 16), acc);
    int n = nt * 16 + (lane & 15);
    float bias = (n < 192) ? q_bias[n] : (n >= 384 ? v_bias[n - 384] : 0.f);
#pragma unroll
    for (int r = 0; r < 8; ++r)
      qkv[(size_t)(win * 64 + mt * 16 + r + 8 * half) * 576 + n] = (_Float16)(acc[r] + bias);
  }
}

// ---------------- attention per (window, head) ----------------
__device__ __forceinline__ int region9(int y, int x) {
  int ry = (y < 56) ? 0 : ((y < 60) ? 1 : 2);
  int rx = (x < 56) ? 0 : ((x < 60) ? 1 : 2);
  return ry * 3 + rx;
}

__global__ void __launch_bounds__(128) k_attn(const _Float16* __restrict__ qkv, const float* __restrict__ biasTab,
                                              const float* __restrict__ logit_scale, _Float16* __restrict__ aout,
                                              int shift) {
  __shared__ float qf[64 * 32];
  __shared__ float kf[64 * 32];
  __shared__ float vf[64 * 32];
  __shared__ float sm[64 * 64];
  int wg = blockIdx.x;
  int win = wg / 6, head = wg % 6;
  int hy = (win >> 3) & 7, wx = win & 7;
  for (int idx = threadIdx.x; idx < 64 * 32; idx += 128) {
    int r = idx >> 5, d = idx & 31;
    size_t base = (size_t)(win * 64 + r) * 576 + head * 32 + d;
    qf[idx] = (float)qkv[base];
    kf[idx] = (float)qkv[base + 192];
    vf[idx] = (float)qkv[base + 384];
  }
  __syncthreads();
  float scale = __expf(fminf(logit_scale[head], LOG100));
  {   // normalize q rows (fold in scale) and k rows: 128 threads -> 128 rows
    int r = threadIdx.x & 63;
    float* buf = (threadIdx.x < 64) ? qf : kf;
    float s = 0.f;
    for (int d = 0; d < 32; ++d) { float v = buf[r * 32 + d]; s += v * v; }
    float inv = 1.f / fmaxf(sqrtf(s), 1e-12f);
    if (threadIdx.x < 64) inv *= scale;
    for (int d = 0; d < 32; ++d) buf[r * 32 + d] *= inv;
  }
  __syncthreads();
  int wv = threadIdx.x >> 5, lane = threadIdx.x & 31, half = lane >> 4;
  // S strip (16 x 64) per wave, one K=32 WMMA per 16x16 tile
  v16h aq = load_a_f(qf, 32, wv * 16, 0);
  for (int jt = 0; jt < 4; ++jt) {
    v8f acc = {0.f,0.f,0.f,0.f,0.f,0.f,0.f,0.f};
    acc = wmma16(aq, load_bT_f(kf, 32, 0, jt * 16), acc);
    int j = jt * 16 + (lane & 15);
    int rj = shift ? region9(hy * 8 + (j >> 3), wx * 8 + (j & 7)) : 0;
#pragma unroll
    for (int r = 0; r < 8; ++r) {
      int i = wv * 16 + r + 8 * half;
      float v = acc[r] + biasTab[head * 4096 + i * 64 + j];
      if (shift) {
        int ri = region9(hy * 8 + (i >> 3), wx * 8 + (i & 7));
        if (ri != rj) v -= 100.f;
      }
      sm[i * 64 + j] = v;
    }
  }
  __syncthreads();
  if (threadIdx.x < 64) {   // row softmax
    int i = threadIdx.x;
    float mx = -1e30f;
    for (int j = 0; j < 64; ++j) mx = fmaxf(mx, sm[i * 64 + j]);
    float s = 0.f;
    for (int j = 0; j < 64; ++j) { float e = __expf(sm[i * 64 + j] - mx); sm[i * 64 + j] = e; s += e; }
    float inv = 1.f / s;
    for (int j = 0; j < 64; ++j) sm[i * 64 + j] *= inv;
  }
  __syncthreads();
  // O strip (16 x 32) = P(16x64) @ V(64x32)
  for (int nt = 0; nt < 2; ++nt) {
    v8f acc = {0.f,0.f,0.f,0.f,0.f,0.f,0.f,0.f};
#pragma unroll
    for (int k0 = 0; k0 < 64; k0 += 32)
      acc = wmma16(load_a_f(sm, 64, wv * 16, k0), load_b_kmaj(vf, 32, k0, nt * 16), acc);
    int n = nt * 16 + (lane & 15);
#pragma unroll
    for (int r = 0; r < 8; ++r)
      aout[(size_t)(win * 64 + wv * 16 + r + 8 * half) * 192 + head * 32 + n] = (_Float16)acc[r];
  }
}

// ---------------- proj GEMM + window-reverse/unroll + LayerNorm + residual ----------------
__global__ void __launch_bounds__(256) k_proj(const _Float16* __restrict__ aout, const _Float16* __restrict__ proj_wh,
                                              const float* __restrict__ proj_b, const float* __restrict__ g,
                                              const float* __restrict__ bvec, const float* __restrict__ xshort,
                                              float* __restrict__ xnext, _Float16* __restrict__ xnexth, int shift) {
  __shared__ _Float16 aA[64 * 192];   // 24 KB
  __shared__ float outb[64 * 192];    // 48 KB
  int win = blockIdx.x;
  int b = win >> 6, hy = (win >> 3) & 7, wx = win & 7;
  const _Float16* src = aout + (size_t)win * 64 * 192;
  for (int idx = threadIdx.x; idx < 1536; idx += 256)
    async_copy16(aA + idx * 8, src + idx * 8);
  async_wait0();
  __syncthreads();
  int wv = threadIdx.x >> 5, lane = threadIdx.x & 31, half = lane >> 4;
  for (int t = wv; t < 48; t += 8) {
    int mt = t / 12, nt = t % 12;
    v8f acc = {0.f,0.f,0.f,0.f,0.f,0.f,0.f,0.f};
#pragma unroll
    for (int k0 = 0; k0 < 192; k0 += 32)
      acc = wmma16(load_a_h(aA, 192, mt * 16, k0), load_bT_h(proj_wh, 192, k0, nt * 16), acc);
    int n = nt * 16 + (lane & 15);
    float bias = proj_b[n];
#pragma unroll
    for (int r = 0; r < 8; ++r)
      outb[(mt * 16 + r + 8 * half) * 192 + n] = acc[r] + bias;
  }
  __syncthreads();
  if (threadIdx.x < 64) {
    int r = threadIdx.x;
    float m = 0.f;
    for (int c = 0; c < 192; ++c) m += outb[r * 192 + c];
    m *= (1.f / 192.f);
    float v = 0.f;
    for (int c = 0; c < 192; ++c) { float d = outb[r * 192 + c] - m; v += d * d; }
    float inv = rsqrtf(v * (1.f / 192.f) + 1e-5f);
    int h = (hy * 8 + (r >> 3) + shift) & 63;
    int w = (wx * 8 + (r & 7) + shift) & 63;
    size_t tok = (size_t)(b * 64 + h) * 64 + w;
    for (int c = 0; c < 192; ++c) {
      float val = xshort[tok * 192 + c] + (outb[r * 192 + c] - m) * inv * g[c] + bvec[c];
      xnext[tok * 192 + c] = val;
      xnexth[tok * 192 + c] = (_Float16)val;
    }
  }
}

// ---------------- fused MLP: fc1 + GELU + fc2 + LayerNorm + residual ----------------
__global__ void __launch_bounds__(256) k_mlp(const float* __restrict__ xin, const _Float16* __restrict__ xinh,
                                             const _Float16* __restrict__ fc1_wh, const float* __restrict__ fc1_b,
                                             const _Float16* __restrict__ fc2_wh, const float* __restrict__ fc2_b,
                                             const float* __restrict__ g, const float* __restrict__ bvec,
                                             float* __restrict__ xnext, _Float16* __restrict__ xnexth) {
  __shared__ _Float16 aA[64 * 192];   // 24 KB : input rows (f16)
  __shared__ _Float16 hb[64 * 384];   // 48 KB : half of the hidden activations (f16)
  __shared__ float outb[64 * 192];    // 48 KB : fc2 output
  int tok0 = blockIdx.x * 64;
  const _Float16* src = xinh + (size_t)tok0 * 192;
  for (int idx = threadIdx.x; idx < 1536; idx += 256)
    async_copy16(aA + idx * 8, src + idx * 8);
  async_wait0();
  __syncthreads();
  int wv = threadIdx.x >> 5, lane = threadIdx.x & 31, half = lane >> 4;
  v8f zero = {0.f,0.f,0.f,0.f,0.f,0.f,0.f,0.f};
  v8f acc2[6];
#pragma unroll
  for (int t = 0; t < 6; ++t) acc2[t] = zero;
  for (int chunk = 0; chunk < 2; ++chunk) {
    // fc1 + GELU for hidden columns [chunk*384, chunk*384+384)
    for (int t = wv; t < 96; t += 8) {
      int mt = t / 24, nt = t % 24;
      int ncol0 = chunk * 384 + nt * 16;
      v8f acc = zero;
#pragma unroll
      for (int k0 = 0; k0 < 192; k0 += 32)
        acc = wmma16(load_a_h(aA, 192, mt * 16, k0), load_bT_h(fc1_wh, 192, k0, ncol0), acc);
      int n = lane & 15;
      float bias = fc1_b[ncol0 + n];
#pragma unroll
      for (int r = 0; r < 8; ++r) {
        float xv = acc[r] + bias;
        float ge = 0.5f * xv * (1.f + erff(xv * 0.70710678118654752f));
        hb[(mt * 16 + r + 8 * half) * 384 + nt * 16 + n] = (_Float16)ge;
      }
    }
    __syncthreads();
    // fc2 partial accumulation over this hidden chunk (register accumulators)
#pragma unroll
    for (int ti = 0; ti < 6; ++ti) {
      int t = wv + ti * 8;
      int mt = t / 12, nt = t % 12;
#pragma unroll
      for (int kk = 0; kk < 384; kk += 32)
        acc2[ti] = wmma16(load_a_h(hb, 384, mt * 16, kk),
                          load_bT_h(fc2_wh, 768, chunk * 384 + kk, nt * 16), acc2[ti]);
    }
    __syncthreads();
  }
#pragma unroll
  for (int ti = 0; ti < 6; ++ti) {
    int t = wv + ti * 8;
    int mt = t / 12, nt = t % 12;
    int n = nt * 16 + (lane & 15);
    float bias = fc2_b[n];
#pragma unroll
    for (int r = 0; r < 8; ++r)
      outb[(mt * 16 + r + 8 * half) * 192 + n] = acc2[ti][r] + bias;
  }
  __syncthreads();
  if (threadIdx.x < 64) {
    int r = threadIdx.x;
    float m = 0.f;
    for (int c = 0; c < 192; ++c) m += outb[r * 192 + c];
    m *= (1.f / 192.f);
    float v = 0.f;
    for (int c = 0; c < 192; ++c) { float d = outb[r * 192 + c] - m; v += d * d; }
    float inv = rsqrtf(v * (1.f / 192.f) + 1e-5f);
    for (int c = 0; c < 192; ++c) {
      float val = xin[(size_t)(tok0 + r) * 192 + c] + (outb[r * 192 + c] - m) * inv * g[c] + bvec[c];
      xnext[(size_t)(tok0 + r) * 192 + c] = val;
      xnexth[(size_t)(tok0 + r) * 192 + c] = (_Float16)val;
    }
  }
}

// ---------------- PatchExpand: GEMM 192->384, pixel shuffle x2, LN(96) ----------------
__global__ void __launch_bounds__(256) k_expand(const _Float16* __restrict__ xinh, const _Float16* __restrict__ exp_wh,
                                                const float* __restrict__ g, const float* __restrict__ bvec,
                                                float* __restrict__ out) {
  __shared__ _Float16 aA[64 * 192];   // 24 KB
  __shared__ float outb[64 * 384];    // 96 KB
  int tok0 = blockIdx.x * 64;
  const _Float16* src = xinh + (size_t)tok0 * 192;
  for (int idx = threadIdx.x; idx < 1536; idx += 256)
    async_copy16(aA + idx * 8, src + idx * 8);
  async_wait0();
  __syncthreads();
  int wv = threadIdx.x >> 5, lane = threadIdx.x & 31, half = lane >> 4;
  for (int t = wv; t < 96; t += 8) {
    int mt = t / 24, nt = t % 24;
    v8f acc = {0.f,0.f,0.f,0.f,0.f,0.f,0.f,0.f};
#pragma unroll
    for (int k0 = 0; k0 < 192; k0 += 32)
      acc = wmma16(load_a_h(aA, 192, mt * 16, k0), load_bT_h(exp_wh, 192, k0, nt * 16), acc);
    int n = nt * 16 + (lane & 15);
#pragma unroll
    for (int r = 0; r < 8; ++r)
      outb[(mt * 16 + r + 8 * half) * 384 + n] = acc[r];
  }
  __syncthreads();
  {   // 256 tasks: 64 rows x 4 chunks of 96 channels; LN then pixel-shuffle scatter
    int r = threadIdx.x >> 2, ch = threadIdx.x & 3;
    const float* rowp = outb + r * 384 + ch * 96;
    float m = 0.f;
    for (int c = 0; c < 96; ++c) m += rowp[c];
    m *= (1.f / 96.f);
    float v = 0.f;
    for (int c = 0; c < 96; ++c) { float d = rowp[c] - m; v += d * d; }
    float inv = rsqrtf(v * (1.f / 96.f) + 1e-5f);
    int tok = tok0 + r;
    int b = tok >> 12, h = (tok >> 6) & 63, w = tok & 63;
    int p = ch >> 1, q = ch & 1;
    size_t opix = ((size_t)(b * 128 + 2 * h + p) * 128 + (2 * w + q)) * 96;
    for (int c = 0; c < 96; ++c)
      out[opix + c] = (rowp[c] - m) * inv * g[c] + bvec[c];
  }
}

extern "C" void kernel_launch(void* const* d_in, const int* in_sizes, int n_in,
                              void* d_out, int out_size, void* d_ws, size_t ws_size,
                              hipStream_t stream) {
  (void)in_sizes; (void)n_in; (void)out_size; (void)ws_size;
  const float* x      = (const float*)d_in[0];
  const float* skip   = (const float*)d_in[1];
  const float* cat_w  = (const float*)d_in[2];
  const float* cat_b  = (const float*)d_in[3];
  const float* qkv_w  = (const float*)d_in[4];
  const float* q_bias = (const float*)d_in[5];
  const float* v_bias = (const float*)d_in[6];
  const float* lscale = (const float*)d_in[7];
  const float* cpb_w1 = (const float*)d_in[8];
  const float* cpb_b1 = (const float*)d_in[9];
  const float* cpb_w2 = (const float*)d_in[10];
  const float* proj_w = (const float*)d_in[11];
  const float* proj_b = (const float*)d_in[12];
  const float* n1g    = (const float*)d_in[13];
  const float* n1b    = (const float*)d_in[14];
  const float* fc1_w  = (const float*)d_in[15];
  const float* fc1_b  = (const float*)d_in[16];
  const float* fc2_w  = (const float*)d_in[17];
  const float* fc2_b  = (const float*)d_in[18];
  const float* n2g    = (const float*)d_in[19];
  const float* n2b    = (const float*)d_in[20];
  const float* exp_w  = (const float*)d_in[21];
  const float* eg     = (const float*)d_in[22];
  const float* eb     = (const float*)d_in[23];

  char* ws = (char*)d_ws;
  float*    xA      = (float*)ws;                         // 50,331,648 B
  float*    xB      = (float*)(ws + 50331648);            // 50,331,648 B
  _Float16* qkvb    = (_Float16*)(ws + 100663296);        // 75,497,472 B
  _Float16* aoutb   = (_Float16*)(ws + 176160768);        // 25,165,824 B
  float*    biasTab = (float*)(ws + 201326592);           //    196,608 B
  float*    cvals   = (float*)(ws + 201523200);           //     10,800 B
  _Float16* xAh     = (_Float16*)(ws + 201534464);        // 25,165,824 B
  _Float16* xBh     = (_Float16*)(ws + 226700288);        // 25,165,824 B
  _Float16* wh      = (_Float16*)(ws + 251866112);        //  2,064,384 B

  // f16 weight mirrors (element offsets within wh)
  _Float16* cat_wh  = wh;                       // 73,728
  _Float16* qkv_wh  = wh + 73728;               // 2 x 110,592
  _Float16* proj_wh = wh + 294912;              // 2 x 36,864
  _Float16* fc1_wh  = wh + 368640;              // 2 x 147,456
  _Float16* fc2_wh  = wh + 663552;              // 2 x 147,456
  _Float16* exp_wh  = wh + 958464;              // 73,728

  k_cvt<<<(73728 + 255) / 256, 256, 0, stream>>>(cat_w, cat_wh, 73728);
  k_cvt<<<(221184 + 255) / 256, 256, 0, stream>>>(qkv_w, qkv_wh, 221184);
  k_cvt<<<(73728 + 255) / 256, 256, 0, stream>>>(proj_w, proj_wh, 73728);
  k_cvt<<<(294912 + 255) / 256, 256, 0, stream>>>(fc1_w, fc1_wh, 294912);
  k_cvt<<<(294912 + 255) / 256, 256, 0, stream>>>(fc2_w, fc2_wh, 294912);
  k_cvt<<<(73728 + 255) / 256, 256, 0, stream>>>(exp_w, exp_wh, 73728);
  k_cpb1<<<(2 * 225 * 6 + 127) / 128, 128, 0, stream>>>(cpb_w1, cpb_b1, cpb_w2, cvals);
  k_cpb2<<<(2 * 6 * 4096) / 128, 128, 0, stream>>>(cvals, biasTab);

  k_fuse<<<NTOK / 64, 256, 0, stream>>>(x, skip, cat_wh, cat_b, xA, xAh);
  for (int i = 0; i < 2; ++i) {
    int shift = (i & 1) ? 4 : 0;
    k_qkv<<<NWIN, 256, 0, stream>>>(xAh, qkv_wh + i * 110592, q_bias + i * 192,
                                    v_bias + i * 192, qkvb, shift);
    k_attn<<<NWIN * HEADS, 128, 0, stream>>>(qkvb, biasTab + i * 6 * 4096, lscale + i * 6,
                                             aoutb, shift);
    k_proj<<<NWIN, 256, 0, stream>>>(aoutb, proj_wh + i * 36864, proj_b + i * 192,
                                     n1g + i * 192, n1b + i * 192, xA, xB, xBh, shift);
    k_mlp<<<NTOK / 64, 256, 0, stream>>>(xB, xBh, fc1_wh + i * 147456, fc1_b + i * 768,
                                         fc2_wh + i * 147456, fc2_b + i * 192,
                                         n2g + i * 192, n2b + i * 192, xA, xAh);
  }
  k_expand<<<NTOK / 64, 256, 0, stream>>>(xAh, exp_wh, eg, eb, (float*)d_out);
}
